// DCMModle_78245714198602
// MI455X (gfx1250) — matrix operand order repaired and checked by reading, not compiled, not run on hardware
//
#include <hip/hip_runtime.h>
#include <hip/hip_bf16.h>
#include <stdint.h>

// ---------------------------------------------------------------------------
// Fused DCM module for MI455X (gfx1250, wave32, WMMA bf16 + TDM store)
//   out = relu(bn3( (relu(bn2( relu(bn1(x@Wred^T)) * gf )) @ Wfus^T) ))
//   gf  = relu(bn_gen( mean_hw(y) @ Wgen^T ))
// ---------------------------------------------------------------------------

typedef float  v8f   __attribute__((ext_vector_type(8)));
typedef __bf16 v16bf __attribute__((ext_vector_type(16)));
typedef unsigned int u32x4 __attribute__((ext_vector_type(4)));
typedef int          i32x4 __attribute__((ext_vector_type(4)));
typedef int          i32x8 __attribute__((ext_vector_type(8)));

union FragBF { v16bf v; uint4 q[2]; };

__device__ __forceinline__ unsigned short f2bf(float f) {
  union { float f; unsigned u; } c; c.f = f;
  unsigned r = c.u + 0x7FFFu + ((c.u >> 16) & 1u);   // round-to-nearest-even
  return (unsigned short)(r >> 16);
}

#define EPS_BN 1e-5f
static constexpr int Bc  = 8;
static constexpr int XCc = 1024;
static constexpr int Cc  = 512;
static constexpr int HWc = 4096;

// LDS map for k_main (all dynamic, no static shared -> base offset 0):
//   [0     .. 5120)   A-stage buffer 0   [64][40] bf16
//   [5120  .. 10240)  A-stage buffer 1   [64][40] bf16
//   [10240 .. 76800)  Tb  intermediate t [64][520] bf16   (GEMM2 A operand)
//   [10240 .. 141312) Os  output stage   [512][64] f32    (reuses Tb region)
static constexpr unsigned kLdsAs0 = 0;
static constexpr unsigned kLdsAs1 = 5120;
static constexpr unsigned kLdsTb  = 10240;
static constexpr unsigned kLdsOs  = 10240;
static constexpr unsigned kLdsTotal = 141312;

// ---------------- kernel 1: global average pool of y -----------------------
__global__ __launch_bounds__(256) void k_pool(const float* __restrict__ y,
                                              float* __restrict__ ypool) {
  __shared__ float red[256];
  const int t = threadIdx.x;
  const float* yb = y + (size_t)blockIdx.x * HWc;
  float s = 0.f;
  #pragma unroll
  for (int i = 0; i < 16; i++) s += yb[t + i * 256];
  red[t] = s;
  __syncthreads();
  for (int st = 128; st > 0; st >>= 1) {
    if (t < st) red[t] += red[t + st];
    __syncthreads();
  }
  if (t == 0) ypool[blockIdx.x] = red[0] * (1.0f / (float)HWc);
}

// ---------------- kernel 2: gf + fused BN coefficients ---------------------
__global__ __launch_bounds__(512) void k_coeff(
    const float* __restrict__ ypool,
    const float* __restrict__ w_gen, const float* __restrict__ b_gen,
    const float* __restrict__ g_gen, const float* __restrict__ be_gen,
    const float* __restrict__ m_gen, const float* __restrict__ v_gen,
    const float* __restrict__ b_red, const float* __restrict__ g_red,
    const float* __restrict__ be_red, const float* __restrict__ m_red,
    const float* __restrict__ v_red,
    const float* __restrict__ g_act, const float* __restrict__ be_act,
    const float* __restrict__ m_act, const float* __restrict__ v_act,
    const float* __restrict__ b_fus, const float* __restrict__ g_fus,
    const float* __restrict__ be_fus, const float* __restrict__ m_fus,
    const float* __restrict__ v_fus,
    float* __restrict__ sg, float* __restrict__ s1, float* __restrict__ o1,
    float* __restrict__ o2, float* __restrict__ s3, float* __restrict__ o3) {
  __shared__ float pl[Cc];
  const int c = threadIdx.x;
  const int b = blockIdx.x;
  pl[c] = ypool[b * Cc + c];
  __syncthreads();
  const float* wg = w_gen + (size_t)c * Cc;
  float dot = 0.f;
  for (int k = 0; k < Cc; k++) dot = fmaf(wg[k], pl[k], dot);
  float ag = g_gen[c] * rsqrtf(v_gen[c] + EPS_BN);
  float gf = fmaxf(ag * (dot + b_gen[c] - m_gen[c]) + be_gen[c], 0.f);
  float s2 = g_act[c] * rsqrtf(v_act[c] + EPS_BN);
  sg[b * Cc + c] = s2 * gf;
  if (b == 0) {
    float a1 = g_red[c] * rsqrtf(v_red[c] + EPS_BN);
    s1[c] = a1;
    o1[c] = a1 * (b_red[c] - m_red[c]) + be_red[c];
    o2[c] = be_act[c] - s2 * m_act[c];
    float a3 = g_fus[c] * rsqrtf(v_fus[c] + EPS_BN);
    s3[c] = a3;
    o3[c] = a3 * (b_fus[c] - m_fus[c]) + be_fus[c];
  }
}

// ---------------- kernel 3: weight -> bf16 WMMA-B swizzle ------------------
// B fragment layout (per 32x16 K x N tile): lane l holds n = l&15,
// k = (i>>3)*16 + (l>>4)*8 + (i&7) for elements i = 0..15; tiles stored contiguously.
__global__ __launch_bounds__(256) void k_swizzle(
    const float* __restrict__ wr, const float* __restrict__ wf,
    unsigned short* __restrict__ wrs, unsigned short* __restrict__ wfs) {
  const int id = blockIdx.x * 256 + threadIdx.x;
  if (id < Cc * XCc) {                      // w_red: 512 x 1024
    int i = id & 15, l = (id >> 4) & 31;
    int tIdx = id >> 9, nt = tIdx & 31, kt = tIdx >> 5;
    int n = nt * 16 + (l & 15);
    int k = kt * 32 + (i >> 3) * 16 + (l >> 4) * 8 + (i & 7);
    wrs[id] = f2bf(wr[(size_t)n * XCc + k]);
  } else {                                  // w_fus: 512 x 512
    int id2 = id - Cc * XCc;
    int i = id2 & 15, l = (id2 >> 4) & 31;
    int tIdx = id2 >> 9, nt = tIdx & 31, kt = tIdx >> 5;
    int n = nt * 16 + (l & 15);
    int k = kt * 32 + (i >> 3) * 16 + (l >> 4) * 8 + (i & 7);
    wfs[id2] = f2bf(wf[(size_t)n * Cc + k]);
  }
}

// ---------------- kernel 4: fused double-GEMM main kernel ------------------
// Block: 64 contiguous pixels within one (b) plane, all 512 channels.
// 8 waves; wave w owns output columns [w*64, w*64+64).
__global__ __launch_bounds__(256) void k_main(
    const float* __restrict__ x,
    const unsigned short* __restrict__ wred,   // swizzled bf16 bits
    const unsigned short* __restrict__ wfus,
    const float* __restrict__ s1, const float* __restrict__ o1,
    const float* __restrict__ sg, const float* __restrict__ o2,
    const float* __restrict__ s3, const float* __restrict__ o3,
    float* __restrict__ out) {
  extern __shared__ char smem[];
  unsigned short* As0 = (unsigned short*)(smem + kLdsAs0);
  unsigned short* As1 = (unsigned short*)(smem + kLdsAs1);
  unsigned short* Tb  = (unsigned short*)(smem + kLdsTb);
  float*          Os  = (float*)(smem + kLdsOs);

  const int tid  = threadIdx.x;
  const int lane = tid & 31;
  const int wave = tid >> 5;
  const int nb   = wave * 64;      // wave's N base
  const int row  = lane & 15;
  const int kh   = lane >> 4;

  const int tile = blockIdx.x;     // 512 tiles, 64 per sample
  const int b    = tile >> 6;
  const int p0   = (tile & 63) * 64;

  const float* xb  = x + (size_t)b * XCc * HWc + p0;
  const float* sgb = sg + b * Cc;

  v8f acc[16];
  #pragma unroll
  for (int i = 0; i < 16; i++)
    #pragma unroll
    for (int j = 0; j < 8; j++) acc[i][j] = 0.f;

  const int kk0 = tid >> 6;   // 0..3
  const int mst = tid & 63;   // pixel within tile

  // ============ GEMM1: A[64x1024](x, bf16) * B[1024x512](Wred^T) ==========
  // Double-buffered A staging: loads for slab kt+1 are issued before the
  // WMMAs of slab kt so HBM latency overlaps matrix work.
  #pragma unroll
  for (int i = 0; i < 8; i++) {
    int kk = i * 4 + kk0;
    As0[mst * 40 + kk] = f2bf(xb[(size_t)kk * HWc + mst]);
  }
  __syncthreads();

  for (int kt = 0; kt < 32; ++kt) {
    unsigned short* Acur = (kt & 1) ? As1 : As0;
    unsigned short* Anxt = (kt & 1) ? As0 : As1;
    const bool hasNext = (kt + 1 < 32);

    float xv[8];
    if (hasNext) {
      #pragma unroll
      for (int i = 0; i < 8; i++) {
        int kk = i * 4 + kk0;
        xv[i] = xb[(size_t)((kt + 1) * 32 + kk) * HWc + mst];
      }
    }
    if (kt + 2 < 32)
      __builtin_prefetch(&xb[(size_t)((kt + 2) * 32 + kk0) * HWc + mst], 0, 0);

    FragBF af[4];
    #pragma unroll
    for (int mi = 0; mi < 4; mi++) {
      const unsigned short* base = Acur + (mi * 16 + row) * 40 + kh * 8;
      af[mi].q[0] = *(const uint4*)(base);        // K = kh*8 .. kh*8+7
      af[mi].q[1] = *(const uint4*)(base + 16);   // K = 16+kh*8 ..
    }
    #pragma unroll
    for (int ni = 0; ni < 4; ni++) {
      FragBF bf;
      const unsigned short* bp =
          wred + ((size_t)(kt * 32 + (nb >> 4) + ni)) * 512 + lane * 16;
      bf.q[0] = *(const uint4*)(bp);
      bf.q[1] = *(const uint4*)(bp + 8);
      #pragma unroll
      for (int mi = 0; mi < 4; mi++)
        acc[mi * 4 + ni] = __builtin_amdgcn_wmma_f32_16x16x32_bf16(
            false, af[mi].v, false, bf.v, (short)0, acc[mi * 4 + ni],
            false, false);
    }

    if (hasNext) {
      #pragma unroll
      for (int i = 0; i < 8; i++) {
        int kk = i * 4 + kk0;
        Anxt[mst * 40 + kk] = f2bf(xv[i]);
      }
    }
    __syncthreads();
  }

  // ===== epilogue1: relu(s1*acc+o1) -> *sg -> relu(+o2) -> bf16 LDS =======
  #pragma unroll
  for (int ni = 0; ni < 4; ni++) {
    const int n = nb + ni * 16 + row;
    const float a1 = s1[n], c1 = o1[n];
    const float a2 = sgb[n], c2 = o2[n];
    #pragma unroll
    for (int mi = 0; mi < 4; mi++) {
      #pragma unroll
      for (int r = 0; r < 8; r++) {
        const int m = mi * 16 + kh * 8 + r;
        float xr = fmaxf(fmaf(a1, acc[mi * 4 + ni][r], c1), 0.f);
        float tv = fmaxf(fmaf(a2, xr, c2), 0.f);
        Tb[m * 520 + n] = f2bf(tv);
        acc[mi * 4 + ni][r] = 0.f;
      }
    }
  }
  __syncthreads();

  // ============ GEMM2: A[64x512](t, LDS) * B[512x512](Wfus^T) =============
  for (int kt = 0; kt < 16; ++kt) {
    FragBF af[4];
    #pragma unroll
    for (int mi = 0; mi < 4; mi++) {
      const unsigned short* base = Tb + (mi * 16 + row) * 520 + kt * 32 + kh * 8;
      af[mi].q[0] = *(const uint4*)(base);
      af[mi].q[1] = *(const uint4*)(base + 16);
    }
    #pragma unroll
    for (int ni = 0; ni < 4; ni++) {
      FragBF bf;
      const unsigned short* bp =
          wfus + ((size_t)(kt * 32 + (nb >> 4) + ni)) * 512 + lane * 16;
      bf.q[0] = *(const uint4*)(bp);
      bf.q[1] = *(const uint4*)(bp + 8);
      #pragma unroll
      for (int mi = 0; mi < 4; mi++)
        acc[mi * 4 + ni] = __builtin_amdgcn_wmma_f32_16x16x32_bf16(
            false, af[mi].v, false, bf.v, (short)0, acc[mi * 4 + ni],
            false, false);
    }
  }

  // ===== epilogue2: relu(s3*acc+o3) -> LDS [512][64] -> TDM tensor store ==
  __syncthreads();   // all GEMM2 LDS reads done before Os overwrites Tb
  #pragma unroll
  for (int ni = 0; ni < 4; ni++) {
    const int n = nb + ni * 16 + row;
    const float a3 = s3[n], c3 = o3[n];
    #pragma unroll
    for (int mi = 0; mi < 4; mi++)
      #pragma unroll
      for (int r = 0; r < 8; r++) {
        const int m = mi * 16 + kh * 8 + r;
        Os[n * 64 + m] = fmaxf(fmaf(a3, acc[mi * 4 + ni][r], c3), 0.f);
      }
  }
  __syncthreads();

  // One TDM per block (wave 0): contiguous LDS tile [512 rows x 64 f32]
  // -> global rows of 256B at stride 16KB (out[b][n][p0..p0+63]).
  if (wave == 0) {
    const uint64_t ga =
        (uint64_t)(uintptr_t)(out + (size_t)b * Cc * HWc + p0);
    u32x4 g0;
    g0[0] = 1u;                                   // count=1, user descriptor
    g0[1] = kLdsOs;                               // lds_addr (bytes)
    g0[2] = (unsigned)(ga & 0xFFFFFFFFu);         // global_addr[31:0]
    g0[3] = (unsigned)((ga >> 32) & 0x01FFFFFFu)  // global_addr[56:32]
            | 0x80000000u;                        // type = 2 ("image")
    i32x8 g1;
    g1[0] = 0x00020000;            // data_size=2 (4B), no mask/pad/iterate
    g1[1] = (int)((HWc & 0xFFFF) << 16);          // tensor_dim0 = 4096 (lo16)
    g1[2] = (int)(((HWc >> 16) & 0xFFFF) |        // tensor_dim0 (hi16)
                  ((Cc & 0xFFFF) << 16));         // tensor_dim1 = 512 (lo16)
    g1[3] = (int)(((Cc >> 16) & 0xFFFF) |         // tensor_dim1 (hi16)
                  (64 << 16));                    // tile_dim0 = 64
    g1[4] = (int)Cc;                              // tile_dim1 = 512, dim2 = 0
    g1[5] = (int)HWc;                             // tensor_dim0_stride lo32
    g1[6] = 0;                                    // stride0 hi16 | stride1 lo
    g1[7] = 0;
    i32x4 g2 = {0, 0, 0, 0};                      // group2 (3D+ / gather) off
    i32x4 g3 = {0, 0, 0, 0};                      // group3 (4D+ / gather) off
    i32x8 gx = {0, 0, 0, 0, 0, 0, 0, 0};          // extra group (6-arg form)
    __builtin_amdgcn_tensor_store_from_lds(g0, g1, g2, g3, gx, 0);
    __builtin_amdgcn_s_wait_tensorcnt(0);
  }
}

// ---------------------------------------------------------------------------
extern "C" void kernel_launch(void* const* d_in, const int* in_sizes, int n_in,
                              void* d_out, int out_size, void* d_ws,
                              size_t ws_size, hipStream_t stream) {
  const float* x      = (const float*)d_in[0];
  const float* y      = (const float*)d_in[1];
  const float* w_red  = (const float*)d_in[2];
  const float* b_red  = (const float*)d_in[3];
  const float* g_red  = (const float*)d_in[4];
  const float* be_red = (const float*)d_in[5];
  const float* m_red  = (const float*)d_in[6];
  const float* v_red  = (const float*)d_in[7];
  const float* w_gen  = (const float*)d_in[8];
  const float* b_gen  = (const float*)d_in[9];
  const float* g_gen  = (const float*)d_in[10];
  const float* be_gen = (const float*)d_in[11];
  const float* m_gen  = (const float*)d_in[12];
  const float* v_gen  = (const float*)d_in[13];
  const float* g_act  = (const float*)d_in[14];
  const float* be_act = (const float*)d_in[15];
  const float* m_act  = (const float*)d_in[16];
  const float* v_act  = (const float*)d_in[17];
  const float* w_fus  = (const float*)d_in[18];
  const float* b_fus  = (const float*)d_in[19];
  const float* g_fus  = (const float*)d_in[20];
  const float* be_fus = (const float*)d_in[21];
  const float* m_fus  = (const float*)d_in[22];
  const float* v_fus  = (const float*)d_in[23];

  // workspace layout (needs ~1.6 MB)
  char* ws = (char*)d_ws;
  float* ypool = (float*)(ws + 0);           // 4096 f
  float* sg    = (float*)(ws + 16384);       // 4096 f
  float* s1    = (float*)(ws + 32768);       // 5 x 512 f
  float* o1 = s1 + 512;
  float* o2 = o1 + 512;
  float* s3 = o2 + 512;
  float* o3 = s3 + 512;
  unsigned short* wredsw = (unsigned short*)(ws + 65536);            // 1 MB
  unsigned short* wfussw = (unsigned short*)(ws + 65536 + 1048576);  // 0.5 MB

  static bool attrDone = false;
  if (!attrDone) {
    (void)hipFuncSetAttribute((const void*)k_main,
                              hipFuncAttributeMaxDynamicSharedMemorySize,
                              (int)kLdsTotal);
    attrDone = true;
  }

  k_pool<<<Bc * Cc, 256, 0, stream>>>(y, ypool);
  k_coeff<<<Bc, 512, 0, stream>>>(ypool, w_gen, b_gen, g_gen, be_gen, m_gen,
                                  v_gen, b_red, g_red, be_red, m_red, v_red,
                                  g_act, be_act, m_act, v_act, b_fus, g_fus,
                                  be_fus, m_fus, v_fus, sg, s1, o1, o2, s3, o3);
  k_swizzle<<<3072, 256, 0, stream>>>(w_red, w_fus, wredsw, wfussw);
  k_main<<<512, 256, kLdsTotal, stream>>>(x, wredsw, wfussw, s1, o1, sg, o2,
                                          s3, o3, (float*)d_out);
  (void)in_sizes; (void)n_in; (void)out_size; (void)ws_size;
}
